// MultiHeadAttention_7292854468646
// MI455X (gfx1250) — matrix operand order repaired and checked by reading, not compiled, not run on hardware
//
#include <hip/hip_runtime.h>

// MHA: B=4, S=2048, E=1024, H=16, D=64. ~137 GFLOP vs <250 MB traffic at
// 23.3 TB/s -> strongly compute bound -> all matmuls via
// v_wmma_f32_16x16x32_bf16. f32->bf16 conversion hoisted to one-shot prepass
// kernels; every hot-loop LDS stage is a pure bf16 async copy in saddr form
// (global_load_async_to_lds_b128 voff, s[base]) with a manually phased double
// buffer: ASYNCcnt retires in order per wave, so s_wait_asynccnt 4 retires
// exactly the previous tile's 4 copies while the next tile's DMA overlaps
// the WMMAs.
//
// Workspace (bf16): Q | Kt | V | O (16 MB each, Kt stored [B,H,D,S]) |
//                   xb (16 MB) | WtQ | WtK | WtV | WtO (2 MB each) = 88 MB.

#define BATCH 4
#define SEQ   2048
#define EMB   1024
#define HEADS 16
#define HDIM  64
#define TOK   (BATCH * SEQ)
#define LOG2E 1.4426950408889634f

typedef __attribute__((ext_vector_type(16))) __bf16 bf16x16;
typedef __attribute__((ext_vector_type(8)))  __bf16 bf16x8;
typedef __attribute__((ext_vector_type(4)))  __bf16 bf16x4;
typedef __attribute__((ext_vector_type(4)))  float  f32x4;
typedef __attribute__((ext_vector_type(8)))  float  f32x8;

union FragU { bf16x16 v; bf16x8 h[2]; };

// ---- CDNA5 async copy (saddr form): 16 B/lane, global -> LDS, ASYNCcnt ----
// mem_addr = SGPR_base + sext(voff); one 32-bit VGPR of address state/lane.
static __device__ inline void async_b128s(unsigned ldsOff, unsigned byteOff,
                                          const void* base) {
  asm volatile("global_load_async_to_lds_b128 %0, %1, %2"
               :: "v"(ldsOff), "v"(byteOff), "s"(base) : "memory");
}
static __device__ inline void async_wait0() {
  asm volatile("s_wait_asynccnt 0x0" ::: "memory");
}
static __device__ inline void async_wait4() {   // keep next tile in flight
  asm volatile("s_wait_asynccnt 0x4" ::: "memory");
}
static __device__ inline unsigned lds_off(const void* p) {
  return (unsigned)(unsigned long long)p;   // low 32 bits of flat shared ptr
}

// A-fragment (16x32, MxK): lane<16 -> row=lane, K={0..7,16..23};
// lane>=16 -> row=lane-16, K={8..15,24..31}.
static __device__ inline bf16x16 load_frag_a(const __bf16* base, int stride) {
  const int lane = threadIdx.x & 31;
  const int m  = lane & 15;
  const int kb = (lane >> 4) << 3;
  FragU f;
  f.h[0] = *(const bf16x8*)(base + m * stride + kb);
  f.h[1] = *(const bf16x8*)(base + m * stride + kb + 16);
  return f.v;
}
// B-fragment (32x16, KxN, [k][n]): lane<16 -> col=lane, K={0..7,16..23};
// lane>=16 -> col=lane-16, K={8..15,24..31}.
static __device__ inline bf16x16 load_frag_b(const __bf16* base, int stride) {
  const int lane = threadIdx.x & 31;
  const int n  = lane & 15;
  const int kb = (lane >> 4) << 3;
  FragU f;
#pragma unroll
  for (int s = 0; s < 8; ++s) {
    f.v[s]     = base[(kb + s) * stride + n];
    f.v[s + 8] = base[(kb + 16 + s) * stride + n];
  }
  return f.v;
}
static __device__ inline f32x8 wmma_bf16(bf16x16 a, bf16x16 b, f32x8 c) {
  return __builtin_amdgcn_wmma_f32_16x16x32_bf16(false, a, false, b, (short)0, c,
                                                 false, false);
}

// ---------------------------------------------------------------------------
// Prepass 1: x (f32 [t][k]) -> xb (bf16 [t][k]), vectorized 4-wide.
// ---------------------------------------------------------------------------
__launch_bounds__(256)
__global__ void cvt_x_kernel(const float* __restrict__ X, __bf16* __restrict__ Y)
{
  size_t i = ((size_t)blockIdx.x * 256 + threadIdx.x) * 4;
  f32x4 v = *(const f32x4*)(X + i);
  bf16x4 o;
  o.x = (__bf16)v.x; o.y = (__bf16)v.y; o.z = (__bf16)v.z; o.w = (__bf16)v.w;
  *(bf16x4*)(Y + i) = o;
}

// ---------------------------------------------------------------------------
// Prepass 2: W (f32 [n][k]) -> Wt (bf16 [k][n]) so GEMM B-tiles are contiguous.
// ---------------------------------------------------------------------------
__launch_bounds__(256)
__global__ void cvt_w_tr_kernel(const float* __restrict__ W, __bf16* __restrict__ Wt)
{
  int idx = blockIdx.x * 256 + threadIdx.x;   // 0 .. EMB*EMB-1
  int n = idx >> 10, k = idx & (EMB - 1);
  Wt[(size_t)k * EMB + n] = (__bf16)W[idx];   // coalesced read, scatter write
}

// ---------------------------------------------------------------------------
// Unified bf16 GEMM:  Y = A[t][k] @ Wt[k][n] + bias[n]
// Block tile 128(M) x 128(N), 8 waves (4x2), 32x64 per wave -> 8 WMMA/k-step.
// mode 0: bf16 out, [B,H,S,D], * scale   (Q / V projections)
// mode 1: f32  out, flat [t][e]          (output projection)
// mode 2: bf16 out, [B,H,D,S]            (K projection, pre-transposed)
// ---------------------------------------------------------------------------
__launch_bounds__(256)
__global__ void gemm_bf16_kernel(const __bf16* __restrict__ A,
                                 const __bf16* __restrict__ Wt,
                                 const float* __restrict__ bias,
                                 float scale, int mode,
                                 __bf16* __restrict__ OutB,
                                 float* __restrict__ OutF)
{
  __shared__ __align__(16) __bf16 sA[2][128 * 32];   // [m][k]
  __shared__ __align__(16) __bf16 sB[2][32 * 128];   // [k][n]
  const int tid  = threadIdx.x;
  const int wave = tid >> 5, lane = tid & 31;
  const int wm = wave >> 1, wn = wave & 1;
  const int m0 = blockIdx.x * 128;
  const int n0 = blockIdx.y * 128;
  const unsigned aB0 = lds_off(sA[0]), aB1 = lds_off(sA[1]);
  const unsigned bB0 = lds_off(sB[0]), bB1 = lds_off(sB[1]);
  const int ia0 = tid, ia1 = tid + 256;
  // byte offsets (32-bit) into A / Wt for this thread's two b128 transfers
  const unsigned aOff0 = ((unsigned)(m0 + (ia0 >> 2)) * EMB + ((ia0 & 3) << 3)) * 2u;
  const unsigned aOff1 = ((unsigned)(m0 + (ia1 >> 2)) * EMB + ((ia1 & 3) << 3)) * 2u;
  const unsigned bOff0 = ((unsigned)(ia0 >> 4) * EMB + n0 + ((ia0 & 15) << 3)) * 2u;
  const unsigned bOff1 = ((unsigned)(ia1 >> 4) * EMB + n0 + ((ia1 & 15) << 3)) * 2u;

  f32x8 acc[2][4] = {};

  auto stage = [&](int k0, unsigned ab, unsigned bb) {
    const unsigned ka = (unsigned)k0 * 2u;            // A advances along k
    const unsigned kb = (unsigned)k0 * (EMB * 2u);    // Wt advances k rows
    async_b128s(ab + ia0 * 16, aOff0 + ka, A);
    async_b128s(ab + ia1 * 16, aOff1 + ka, A);
    async_b128s(bb + ia0 * 16, bOff0 + kb, Wt);
    async_b128s(bb + ia1 * 16, bOff1 + kb, Wt);
  };
  auto compute = [&](const __bf16* pa, const __bf16* pb) {
    bf16x16 a0 = load_frag_a(&pa[(wm * 32) * 32], 32);
    bf16x16 a1 = load_frag_a(&pa[(wm * 32 + 16) * 32], 32);
#pragma unroll
    for (int nt = 0; nt < 4; ++nt) {
      bf16x16 bf = load_frag_b(&pb[wn * 64 + nt * 16], 128);
      acc[0][nt] = wmma_bf16(a0, bf, acc[0][nt]);
      acc[1][nt] = wmma_bf16(a1, bf, acc[1][nt]);
    }
  };

  stage(0, aB0, bB0);
  for (int k0 = 0; k0 < EMB; k0 += 64) {   // two phased k-steps per body
    if (k0 + 32 < EMB) { stage(k0 + 32, aB1, bB1); async_wait4(); }
    else               { async_wait0(); }
    __syncthreads();
    compute(sA[0], sB[0]);
    __syncthreads();
    if (k0 + 64 < EMB) { stage(k0 + 64, aB0, bB0); async_wait4(); }
    else               { async_wait0(); }
    __syncthreads();
    compute(sA[1], sB[1]);
    __syncthreads();
  }

#pragma unroll
  for (int mt = 0; mt < 2; ++mt)
#pragma unroll
    for (int nt = 0; nt < 4; ++nt) {
      int col = n0 + wn * 64 + nt * 16 + (lane & 15);
      float bv = bias[col];
      int mb = m0 + wm * 32 + mt * 16 + ((lane >> 4) << 3);
      if (mode == 1) {
#pragma unroll
        for (int i = 0; i < 8; ++i)
          OutF[(size_t)(mb + i) * EMB + col] = acc[mt][nt][i] + bv;
      } else if (mode == 0) {
        int h = col >> 6, d = col & 63;
#pragma unroll
        for (int i = 0; i < 8; ++i) {
          int t = mb + i, b = t >> 11, s = t & 2047;
          OutB[(((size_t)b * HEADS + h) * SEQ + s) * HDIM + d] =
              (__bf16)((acc[mt][nt][i] + bv) * scale);
        }
      } else {  // mode 2: K pre-transposed [B,H,D,S]
        int h = col >> 6, d = col & 63;
#pragma unroll
        for (int i = 0; i < 8; ++i) {
          int t = mb + i, b = t >> 11, s = t & 2047;
          OutB[(((size_t)b * HEADS + h) * HDIM + d) * SEQ + s] =
              (__bf16)(acc[mt][nt][i] + bv);
        }
      }
    }
}

// ---------------------------------------------------------------------------
// Flash attention per (b,h, 128-query block). 8 waves x 16 query rows.
// Scores produced in the exp2 domain (Q pre-scaled by log2(e)/sqrt(D),
// mask scaled by log2(e) on load) so softmax uses native v_exp_f32 (exp2f).
// K arrives pre-transposed [B,H,D,S]; K/V tiles phased double-buffer async.
// ---------------------------------------------------------------------------
__launch_bounds__(256)
__global__ void attn_kernel(const __bf16* __restrict__ Q, const __bf16* __restrict__ Kt,
                            const __bf16* __restrict__ V, const float* __restrict__ mask,
                            __bf16* __restrict__ O)
{
  __shared__ __align__(16) __bf16 sQ[128 * 64];        // [q][d]
  __shared__ __align__(16) __bf16 sK[2][64 * 64];      // [d][key]
  __shared__ __align__(16) __bf16 sV[2][64 * 64];      // [key][d]
  __shared__ __align__(16) __bf16 sP[8 * 16 * 64];     // per-wave P relayout
  const int tid = threadIdx.x, wave = tid >> 5, lane = tid & 31;
  const int bh = blockIdx.y;
  const int b = bh >> 4, h = bh & 15;
  const int q0 = blockIdx.x * 128;
  const __bf16* Qb  = Q  + ((size_t)bh * SEQ + q0) * HDIM;   // contiguous tile
  const __bf16* Ktb = Kt + (size_t)bh * HDIM * SEQ;          // [d][s]
  const __bf16* Vb  = V  + (size_t)bh * SEQ * HDIM;          // [s][d]
  const unsigned qBase = lds_off(sQ);
  const unsigned kB0 = lds_off(sK[0]), kB1 = lds_off(sK[1]);
  const unsigned vB0 = lds_off(sV[0]), vB1 = lds_off(sV[1]);
  const int i0 = tid, i1 = tid + 256;
  // 32-bit byte offsets for this thread's transfers
  const unsigned kOff0 = ((unsigned)(i0 >> 3) * SEQ + ((i0 & 7) << 3)) * 2u;
  const unsigned kOff1 = ((unsigned)(i1 >> 3) * SEQ + ((i1 & 7) << 3)) * 2u;
  const unsigned vOff0 = (unsigned)i0 * 16u;
  const unsigned vOff1 = (unsigned)i1 * 16u;

  auto stageKV = [&](int kb, unsigned kbase, unsigned vbase) {
    const unsigned kByte = (unsigned)kb * 2u;            // key advance in Kt row
    const unsigned vByte = (unsigned)kb * (HDIM * 2u);   // row advance in V
    async_b128s(kbase + i0 * 16, kOff0 + kByte, Ktb);
    async_b128s(kbase + i1 * 16, kOff1 + kByte, Ktb);
    async_b128s(vbase + i0 * 16, vOff0 + vByte, Vb);
    async_b128s(vbase + i1 * 16, vOff1 + vByte, Vb);
  };

  f32x8 oacc[4] = {};
  float mrun[8], lrun[8];
#pragma unroll
  for (int i = 0; i < 8; ++i) { mrun[i] = -1e30f; lrun[i] = 0.f; }

  __bf16* pw = &sP[wave * 16 * 64];
  bf16x16 qa[2];

  auto attn_block = [&](const __bf16* pk, const __bf16* pv, int kb) {
    // S = Q K^T  (16 x 64), K-dim = HDIM = 64 -> two 32-wide k-steps
    f32x8 sc[4] = {};
#pragma unroll
    for (int c = 0; c < 2; ++c)
#pragma unroll
      for (int j = 0; j < 4; ++j) {
        bf16x16 kf = load_frag_b(&pk[(c * 32) * 64 + j * 16], 64);
        sc[j] = wmma_bf16(qa[c], kf, sc[j]);
      }
    // additive mask (broadcast over queries), in exp2 domain
#pragma unroll
    for (int j = 0; j < 4; ++j) {
      float mk = mask[(size_t)b * SEQ + kb + j * 16 + (lane & 15)] * LOG2E;
#pragma unroll
      for (int i = 0; i < 8; ++i) sc[j][i] += mk;
    }
    // online softmax (base-2): rows = VGPR slots, cols across 16-lane halves
#pragma unroll
    for (int i = 0; i < 8; ++i) {
      float v = fmaxf(fmaxf(sc[0][i], sc[1][i]), fmaxf(sc[2][i], sc[3][i]));
      v = fmaxf(v, __shfl_xor(v, 1, 32));
      v = fmaxf(v, __shfl_xor(v, 2, 32));
      v = fmaxf(v, __shfl_xor(v, 4, 32));
      v = fmaxf(v, __shfl_xor(v, 8, 32));
      float mnew = fmaxf(mrun[i], v);
      float corr = exp2f(mrun[i] - mnew);
      mrun[i] = mnew;
      float rs = 0.f;
#pragma unroll
      for (int j = 0; j < 4; ++j) { sc[j][i] = exp2f(sc[j][i] - mnew); rs += sc[j][i]; }
      rs += __shfl_xor(rs, 1, 32);
      rs += __shfl_xor(rs, 2, 32);
      rs += __shfl_xor(rs, 4, 32);
      rs += __shfl_xor(rs, 8, 32);
      lrun[i] = lrun[i] * corr + rs;
#pragma unroll
      for (int j = 0; j < 4; ++j) oacc[j][i] *= corr;
    }
    // P: C/D layout -> A layout via per-wave LDS scratch (wave-private)
    const int rb = (lane >> 4) << 3;
    const int cn = lane & 15;
#pragma unroll
    for (int j = 0; j < 4; ++j)
#pragma unroll
      for (int i = 0; i < 8; ++i)
        pw[(rb + i) * 64 + j * 16 + cn] = (__bf16)sc[j][i];
    // O += P V   (16 x 64), K-dim = 64 keys -> two k-steps
#pragma unroll
    for (int c = 0; c < 2; ++c) {
      bf16x16 pa = load_frag_a(&pw[c * 32], 64);
#pragma unroll
      for (int j = 0; j < 4; ++j) {
        bf16x16 vf = load_frag_b(&pv[(c * 32) * 64 + j * 16], 64);
        oacc[j] = wmma_bf16(pa, vf, oacc[j]);
      }
    }
  };

  // stage Q (contiguous 16 KB) + first K/V tile; retire Q (in-order: first 4)
#pragma unroll
  for (int i = 0; i < 4; ++i) {
    int idx = i * 256 + tid;
    async_b128s(qBase + idx * 16, (unsigned)idx * 16u, Qb);
  }
  stageKV(0, kB0, vB0);
  async_wait4();
  __syncthreads();
  qa[0] = load_frag_a(&sQ[wave * 16 * 64], 64);
  qa[1] = load_frag_a(&sQ[wave * 16 * 64 + 32], 64);

  for (int kb = 0; kb < SEQ; kb += 128) {   // two phased key blocks per body
    if (kb + 64 < SEQ) { stageKV(kb + 64, kB1, vB1); async_wait4(); }
    else               { async_wait0(); }
    __syncthreads();
    attn_block(sK[0], sV[0], kb);
    __syncthreads();
    if (kb + 128 < SEQ) { stageKV(kb + 128, kB0, vB0); async_wait4(); }
    else               { async_wait0(); }
    __syncthreads();
    attn_block(sK[1], sV[1], kb + 64);
    __syncthreads();
  }

  // normalize, write O as [B,S,H,D] bf16 (== [t][e] for the output projection)
#pragma unroll
  for (int j = 0; j < 4; ++j) {
    int d = j * 16 + (lane & 15);
    int rb = (lane >> 4) << 3;
#pragma unroll
    for (int i = 0; i < 8; ++i) {
      int s = q0 + wave * 16 + rb + i;
      O[(((size_t)b * SEQ + s) * HEADS + h) * HDIM + d] =
          (__bf16)(oacc[j][i] / lrun[i]);
    }
  }
}

// ---------------------------------------------------------------------------
extern "C" void kernel_launch(void* const* d_in, const int* in_sizes, int n_in,
                              void* d_out, int out_size, void* d_ws, size_t ws_size,
                              hipStream_t stream)
{
  (void)in_sizes; (void)n_in; (void)out_size; (void)ws_size;
  const float* x    = (const float*)d_in[0];
  const float* mask = (const float*)d_in[1];
  const float* Wq   = (const float*)d_in[2];
  const float* bq   = (const float*)d_in[3];
  const float* Wk   = (const float*)d_in[4];
  const float* bk   = (const float*)d_in[5];
  const float* Wv   = (const float*)d_in[6];
  const float* bv   = (const float*)d_in[7];
  const float* Wo   = (const float*)d_in[8];
  const float* bo   = (const float*)d_in[9];
  float* out = (float*)d_out;

  const size_t tokE = (size_t)TOK * EMB;       // 8,388,608
  const size_t wE   = (size_t)EMB * EMB;       // 1,048,576
  __bf16* Qw  = (__bf16*)d_ws;
  __bf16* Ktw = Qw  + tokE;
  __bf16* Vw  = Ktw + tokE;
  __bf16* Ow  = Vw  + tokE;
  __bf16* xb  = Ow  + tokE;
  __bf16* WtQ = xb  + tokE;
  __bf16* WtK = WtQ + wE;
  __bf16* WtV = WtK + wE;
  __bf16* WtO = WtV + wE;                      // total 88 MB

  dim3 blk(256);
  // Prepass: convert/transposes (tiny vs 137 GFLOP of WMMA work)
  cvt_x_kernel<<<dim3(tokE / 1024), blk, 0, stream>>>(x, xb);
  cvt_w_tr_kernel<<<dim3(wE / 256), blk, 0, stream>>>(Wq, WtQ);
  cvt_w_tr_kernel<<<dim3(wE / 256), blk, 0, stream>>>(Wk, WtK);
  cvt_w_tr_kernel<<<dim3(wE / 256), blk, 0, stream>>>(Wv, WtV);
  cvt_w_tr_kernel<<<dim3(wE / 256), blk, 0, stream>>>(Wo, WtO);

  dim3 gGemm(TOK / 128, EMB / 128);            // 64 x 8
  const float qscale = 0.125f * LOG2E;         // 1/sqrt(D) * log2(e)
  gemm_bf16_kernel<<<gGemm, blk, 0, stream>>>(xb, WtQ, bq, qscale, 0, Qw, nullptr);
  gemm_bf16_kernel<<<gGemm, blk, 0, stream>>>(xb, WtK, bk, 1.0f,   2, Ktw, nullptr);
  gemm_bf16_kernel<<<gGemm, blk, 0, stream>>>(xb, WtV, bv, 1.0f,   0, Vw, nullptr);

  dim3 gAttn(SEQ / 128, BATCH * HEADS);        // 16 x 64
  attn_kernel<<<gAttn, blk, 0, stream>>>(Qw, Ktw, Vw, mask, Ow);

  gemm_bf16_kernel<<<gGemm, blk, 0, stream>>>(Ow, WtO, bo, 1.0f, 1, nullptr, out);
}
